// SwinTransformer_20959440404588
// MI455X (gfx1250) — matrix operand order repaired and checked by reading, not compiled, not run on hardware
//
#include <hip/hip_runtime.h>

typedef __attribute__((ext_vector_type(16))) __bf16 v16bf;
typedef __attribute__((ext_vector_type(8)))  float  v8f;
typedef int v4i __attribute__((vector_size(4 * sizeof(int))));

typedef __attribute__((address_space(1))) v4i g_v4i;   // global int4
typedef __attribute__((address_space(3))) v4i l_v4i;   // LDS int4

#define HW_    28
#define C_     256
#define NTOK   49
#define HEADS_ 8
#define HD_    32

#if __has_builtin(__builtin_amdgcn_global_load_async_to_lds_b128)
#define ASYNC_LDS 1
#else
#define ASYNC_LDS 0
#endif

union Frag {
    v16bf v;
    uint4 q[2];
};

__device__ __forceinline__ v8f wmma_bf16(const Frag& a, const Frag& b, v8f c) {
    return __builtin_amdgcn_wmma_f32_16x16x32_bf16(false, a.v, false, b.v,
                                                   (short)0, c, false, false);
}

__device__ __forceinline__ void wait_async0() {
#if __has_builtin(__builtin_amdgcn_s_wait_asynccnt)
    __builtin_amdgcn_s_wait_asynccnt(0);
#else
    asm volatile("s_wait_asynccnt 0x0" ::: "memory");
#endif
}

// ---------------------------------------------------------------------------
// Weight convert + transpose: f32 (K,N) -> bf16 (N,K) so WMMA B-fragments are
// two 16B loads per lane (lanes 0-15: K=0-7,16-23, lanes 16-31: K=8-15,24-31).
// ---------------------------------------------------------------------------
__global__ void wconv_kernel(const float* __restrict__ w, __bf16* __restrict__ wt,
                             int K, int N) {
    int i = blockIdx.x * 256 + threadIdx.x;
    if (i < K * N) {
        int k = i / N, n = i % N;
        wt[(size_t)n * K + k] = (__bf16)w[i];
    }
}

// ---------------------------------------------------------------------------
// LayerNorm over C=256, one wave32 per token, 8 channels per lane.
// ---------------------------------------------------------------------------
__global__ __launch_bounds__(256) void ln_kernel(const float* __restrict__ x,
                                                 const float* __restrict__ g,
                                                 const float* __restrict__ b,
                                                 __bf16* __restrict__ y,
                                                 int tokens) {
    int row  = blockIdx.x * 8 + (threadIdx.x >> 5);
    int lane = threadIdx.x & 31;
    if (row >= tokens) return;
    const float* xr = x + (size_t)row * C_ + lane * 8;
    float4 v0 = *(const float4*)xr;
    float4 v1 = *(const float4*)(xr + 4);
    float vv[8] = {v0.x, v0.y, v0.z, v0.w, v1.x, v1.y, v1.z, v1.w};
    float s = 0.f, ss = 0.f;
#pragma unroll
    for (int e = 0; e < 8; e++) { s += vv[e]; ss += vv[e] * vv[e]; }
#pragma unroll
    for (int off = 16; off >= 1; off >>= 1) {
        s  += __shfl_xor(s,  off, 32);
        ss += __shfl_xor(ss, off, 32);
    }
    float mu  = s * (1.0f / C_);
    float var = ss * (1.0f / C_) - mu * mu;
    float rs  = rsqrtf(var + 1.001e-5f);
    const float* gp = g + lane * 8;
    const float* bp = b + lane * 8;
    union { uint4 u; __bf16 h[8]; } o;
#pragma unroll
    for (int e = 0; e < 8; e++)
        o.h[e] = (__bf16)((vv[e] - mu) * rs * gp[e] + bp[e]);
    *(uint4*)(y + (size_t)row * C_ + lane * 8) = o.u;
}

// ---------------------------------------------------------------------------
// Tiled WMMA GEMM:  out = A(MxK bf16) * Wt^T(N,K bf16) + bias  [+ epilogue]
//   EP=0: bias -> bf16   EP=1: bias + f32 residual -> f32
//   EP=2: bias + exact GELU -> bf16
// Block tile 128x64, 8 waves x (2x2) 16x16x32 WMMA tiles, BK=32.
// Double-buffered LDS; K-step i+1 prefetched with GLOBAL_LOAD_ASYNC_TO_LDS
// (ASYNCcnt) while step i runs on the matrix pipe.
// ---------------------------------------------------------------------------
template <int EP>
__global__ __launch_bounds__(256) void gemm_kernel(const __bf16* __restrict__ A,
                                                   const __bf16* __restrict__ Wt,
                                                   const float* __restrict__ bias,
                                                   const float* __restrict__ res,
                                                   void* __restrict__ outp,
                                                   int M, int N, int K) {
    (void)M;
    __shared__ __attribute__((aligned(32))) __bf16 As[2 * 128 * 40]; // row stride 40
    __shared__ __attribute__((aligned(32))) __bf16 Bs[2 * 64 * 48];  // row stride 48

    const int tid  = threadIdx.x;
    const int m0   = blockIdx.y * 128;
    const int n0   = blockIdx.x * 64;
    const int wv   = tid >> 5, lane = tid & 31;
    const int wm   = (wv & 3) * 32, wn = (wv >> 2) * 32;
    const int lhi  = lane >> 4, llo = lane & 15;

    const v8f z = {0.f, 0.f, 0.f, 0.f, 0.f, 0.f, 0.f, 0.f};
    v8f acc[2][2];
    acc[0][0] = z; acc[0][1] = z; acc[1][0] = z; acc[1][1] = z;

    const int brow = tid >> 2, bseg = tid & 3;   // 64 rows x 4 segs of 8 (B tile)

#if ASYNC_LDS
    auto issue = [&](int kb, int buf) {
        __bf16* Ab = As + buf * (128 * 40);
        __bf16* Bb = Bs + buf * (64 * 48);
#pragma unroll
        for (int r = 0; r < 2; r++) {
            int c   = tid + 256 * r;             // 512 16B chunks of the A tile
            int row = c >> 2, seg = c & 3;
            __builtin_amdgcn_global_load_async_to_lds_b128(
                (g_v4i*)(A + (size_t)(m0 + row) * K + kb + seg * 8),
                (l_v4i*)(Ab + row * 40 + seg * 8), 0, 0);
        }
        __builtin_amdgcn_global_load_async_to_lds_b128(
            (g_v4i*)(Wt + (size_t)(n0 + brow) * K + kb + bseg * 8),
            (l_v4i*)(Bb + brow * 48 + bseg * 8), 0, 0);
    };
    issue(0, 0);
#else
    const int arow = tid >> 1, ahalf = tid & 1;  // 128 rows x 2 halves of 16
#endif

    int buf = 0;
    for (int kb = 0; kb < K; kb += 32, buf ^= 1) {
#if ASYNC_LDS
        wait_async0();
        __syncthreads();
        if (kb + 32 < K) issue(kb + 32, buf ^ 1);
        const __bf16* Ab = As + buf * (128 * 40);
        const __bf16* Bb = Bs + buf * (64 * 48);
#else
        __bf16* Ab = As + buf * (128 * 40);
        __bf16* Bb = Bs + buf * (64 * 48);
        {
            const uint4* ga = (const uint4*)(A + (size_t)(m0 + arow) * K + kb + ahalf * 16);
            uint4* da = (uint4*)(Ab + arow * 40 + ahalf * 16);
            da[0] = ga[0];
            da[1] = ga[1];
            *(uint4*)(Bb + brow * 48 + bseg * 8) =
                *(const uint4*)(Wt + (size_t)(n0 + brow) * K + kb + bseg * 8);
        }
        __syncthreads();
#endif

        Frag a[2], b[2];
#pragma unroll
        for (int i = 0; i < 2; i++) {
            int r = wm + i * 16 + llo;
            a[i].q[0] = *(const uint4*)(Ab + r * 40 + lhi * 8);
            a[i].q[1] = *(const uint4*)(Ab + r * 40 + 16 + lhi * 8);
        }
#pragma unroll
        for (int j = 0; j < 2; j++) {
            int r = wn + j * 16 + llo;
            b[j].q[0] = *(const uint4*)(Bb + r * 48 + lhi * 8);
            b[j].q[1] = *(const uint4*)(Bb + r * 48 + 16 + lhi * 8);
        }
#pragma unroll
        for (int i = 0; i < 2; i++)
#pragma unroll
            for (int j = 0; j < 2; j++)
                acc[i][j] = wmma_bf16(a[i], b[j], acc[i][j]);
#if !ASYNC_LDS
        __syncthreads();
#endif
    }

#pragma unroll
    for (int i = 0; i < 2; i++) {
#pragma unroll
        for (int j = 0; j < 2; j++) {
#pragma unroll
            for (int r = 0; r < 8; r++) {
                int m = m0 + wm + i * 16 + lhi * 8 + r;
                int n = n0 + wn + j * 16 + llo;
                float v = acc[i][j][r] + bias[n];
                size_t idx = (size_t)m * N + n;
                if (EP == 1) {
                    ((float*)outp)[idx] = v + res[idx];
                } else if (EP == 2) {
                    float gz = 0.5f * v * (1.0f + erff(v * 0.70710678118654752f));
                    ((__bf16*)outp)[idx] = (__bf16)gz;
                } else {
                    ((__bf16*)outp)[idx] = (__bf16)v;
                }
            }
        }
    }
}

// ---------------------------------------------------------------------------
// Windowed attention, one block = one (window, head); 128 threads (4 waves).
// The additive (rel-bias + shift-mask) term is precomputed into the score
// LDS buffer during the gather phase, so the post-WMMA epilogue is a
// branch-free ds_load + fma + ds_store.
// ---------------------------------------------------------------------------
__global__ __launch_bounds__(128) void attn_kernel(const __bf16* __restrict__ qkv,
                                                   const float* __restrict__ bias_tab,
                                                   __bf16* __restrict__ attn,
                                                   int shift) {
    __shared__ __attribute__((aligned(32))) __bf16 q_s[64 * 40];
    __shared__ __attribute__((aligned(32))) __bf16 k_s[64 * 48];
    __shared__ __attribute__((aligned(32))) __bf16 v_t[32 * 80];  // v transposed (d, key)
    __shared__ __attribute__((aligned(32))) float  s_s[64 * 65];
    __shared__ __attribute__((aligned(32))) __bf16 p_s[64 * 72];

    const int tid  = threadIdx.x;
    const int head = blockIdx.y;
    const int wid  = blockIdx.x;
    const int img  = wid >> 4;
    const int win  = wid & 15;
    const int wi   = win >> 2, wj = win & 3;

    // ---- gather q, k, v (64 rows x 2 half-threads, 16 channels each) ----
    {
        const int n = tid >> 1, half = tid & 1;
        if (n < NTOK) {
            int a = n / 7, c = n % 7;
            int h = wi * 7 + a + shift; if (h >= HW_) h -= HW_;
            int w = wj * 7 + c + shift; if (w >= HW_) w -= HW_;
            size_t t = (size_t)(img * HW_ + h) * HW_ + w;
            const __bf16* base = qkv + t * 768 + head * HD_ + half * 16;
            const uint4* pq = (const uint4*)(base);
            const uint4* pk = (const uint4*)(base + 256);
            const uint4* pv = (const uint4*)(base + 512);
            *(uint4*)(q_s + n * 40 + half * 16)     = pq[0];
            *(uint4*)(q_s + n * 40 + half * 16 + 8) = pq[1];
            *(uint4*)(k_s + n * 48 + half * 16)     = pk[0];
            *(uint4*)(k_s + n * 48 + half * 16 + 8) = pk[1];
            union { uint4 u[2]; __bf16 h16[16]; } vv;
            vv.u[0] = pv[0];
            vv.u[1] = pv[1];
#pragma unroll
            for (int e = 0; e < 16; e++)
                v_t[(half * 16 + e) * 80 + n] = vv.h16[e];
        } else {
            uint4 zz = {0u, 0u, 0u, 0u};
            *(uint4*)(q_s + n * 40 + half * 16)     = zz;
            *(uint4*)(q_s + n * 40 + half * 16 + 8) = zz;
            *(uint4*)(k_s + n * 48 + half * 16)     = zz;
            *(uint4*)(k_s + n * 48 + half * 16 + 8) = zz;
#pragma unroll
            for (int e = 0; e < 16; e++)
                v_t[(half * 16 + e) * 80 + n] = (__bf16)0.0f;
        }
    }

    // ---- precompute additive term (relative bias + shift mask) ----
    for (int idx = tid; idx < 64 * 64; idx += 128) {
        int m  = idx >> 6, nn = idx & 63;
        float val = 0.0f;
        if (m < NTOK && nn < NTOK) {
            int fx = m / 7, fy = m % 7, gx = nn / 7, gy = nn % 7;
            val = bias_tab[((fx - gx + 6) * 13 + (fy - gy + 6)) * HEADS_ + head];
            if (shift) {
                int rm = wi * 7 + fx, cm = wj * 7 + fy;
                int rn = wi * 7 + gx, cn = wj * 7 + gy;
                int am = (rm < 21 ? 0 : (rm < 25 ? 1 : 2)) * 3 +
                         (cm < 21 ? 0 : (cm < 25 ? 1 : 2));
                int an = (rn < 21 ? 0 : (rn < 25 ? 1 : 2)) * 3 +
                         (cn < 21 ? 0 : (cn < 25 ? 1 : 2));
                if (am != an) val -= 100.0f;
            }
        }
        s_s[m * 65 + nn] = val;
    }
    __syncthreads();

    const int wv  = tid >> 5, lane = tid & 31;
    const int lhi = lane >> 4, llo = lane & 15;
    const v8f z = {0.f, 0.f, 0.f, 0.f, 0.f, 0.f, 0.f, 0.f};

    // ---- scores: wave wv owns row tile m = wv*16 .. wv*16+15 ----
    {
        Frag a;
        int r = wv * 16 + llo;
        a.q[0] = *(const uint4*)(q_s + r * 40 + lhi * 8);
        a.q[1] = *(const uint4*)(q_s + r * 40 + 16 + lhi * 8);
        const float scale = 0.17677669529663687f;  // 1/sqrt(32)
#pragma unroll
        for (int j = 0; j < 4; j++) {
            Frag b;
            int kr = j * 16 + llo;
            b.q[0] = *(const uint4*)(k_s + kr * 48 + lhi * 8);
            b.q[1] = *(const uint4*)(k_s + kr * 48 + 16 + lhi * 8);
            v8f cs = wmma_bf16(a, b, z);
#pragma unroll
            for (int rr = 0; rr < 8; rr++) {
                int m  = wv * 16 + lhi * 8 + rr;
                int nn = j * 16 + llo;
                int o  = m * 65 + nn;
                s_s[o] = cs[rr] * scale + s_s[o];
            }
        }
    }
    __syncthreads();

    // ---- softmax: one thread per row; bf16 probs (cols>=49 zeroed) ----
    if (tid < 64) {
        float mx = -1e30f;
        for (int c = 0; c < NTOK; c++) mx = fmaxf(mx, s_s[tid * 65 + c]);
        float sum = 0.f;
        for (int c = 0; c < NTOK; c++) sum += __expf(s_s[tid * 65 + c] - mx);
        float inv = 1.0f / sum;
        for (int c = 0; c < 64; c++) {
            float pv = (c < NTOK) ? __expf(s_s[tid * 65 + c] - mx) * inv : 0.0f;
            p_s[tid * 72 + c] = (__bf16)pv;
        }
    }
    __syncthreads();

    // ---- context: out = P(64x64) @ V(64x32), K in two 32-steps ----
    {
        v8f co[2];
        co[0] = z; co[1] = z;
#pragma unroll
        for (int kb = 0; kb < 64; kb += 32) {
            Frag a;
            int r = wv * 16 + llo;
            a.q[0] = *(const uint4*)(p_s + r * 72 + kb + lhi * 8);
            a.q[1] = *(const uint4*)(p_s + r * 72 + kb + 16 + lhi * 8);
#pragma unroll
            for (int j = 0; j < 2; j++) {
                Frag b;
                int dr = j * 16 + llo;
                b.q[0] = *(const uint4*)(v_t + dr * 80 + kb + lhi * 8);
                b.q[1] = *(const uint4*)(v_t + dr * 80 + kb + 16 + lhi * 8);
                co[j] = wmma_bf16(a, b, co[j]);
            }
        }
#pragma unroll
        for (int j = 0; j < 2; j++) {
#pragma unroll
            for (int rr = 0; rr < 8; rr++) {
                int m = wv * 16 + lhi * 8 + rr;
                if (m < NTOK) {
                    int a2 = m / 7, c2 = m % 7;
                    int h = wi * 7 + a2 + shift; if (h >= HW_) h -= HW_;
                    int w = wj * 7 + c2 + shift; if (w >= HW_) w -= HW_;
                    size_t t = (size_t)(img * HW_ + h) * HW_ + w;
                    attn[t * C_ + head * HD_ + j * 16 + llo] = (__bf16)co[j][rr];
                }
            }
        }
    }
}

// ---------------------------------------------------------------------------
// Host orchestration
// ---------------------------------------------------------------------------
extern "C" void kernel_launch(void* const* d_in, const int* in_sizes, int n_in,
                              void* d_out, int out_size, void* d_ws, size_t ws_size,
                              hipStream_t stream) {
    (void)in_sizes; (void)n_in; (void)out_size; (void)ws_size;

    const int Bimg = 64, NCH = 4;
    const int IMGC = Bimg / NCH;               // 16 images per chunk
    const int TPC  = IMGC * HW_ * HW_;         // 12544 tokens per chunk
    const int TOK  = Bimg * HW_ * HW_;         // 50176 tokens total

    const float* x = (const float*)d_in[0];
    struct BlkP {
        const float *ln1g, *ln1b, *qkvw, *qkvb, *btab, *projw, *projb;
        const float *ln2g, *ln2b, *fc1w, *fc1b, *fc2w, *fc2b;
    } bp[2];
    for (int b = 0; b < 2; b++) {
        int o = 1 + 13 * b;
        bp[b].ln1g  = (const float*)d_in[o + 0];
        bp[b].ln1b  = (const float*)d_in[o + 1];
        bp[b].qkvw  = (const float*)d_in[o + 2];
        bp[b].qkvb  = (const float*)d_in[o + 3];
        bp[b].btab  = (const float*)d_in[o + 4];
        bp[b].projw = (const float*)d_in[o + 5];
        bp[b].projb = (const float*)d_in[o + 6];
        bp[b].ln2g  = (const float*)d_in[o + 7];
        bp[b].ln2b  = (const float*)d_in[o + 8];
        bp[b].fc1w  = (const float*)d_in[o + 9];
        bp[b].fc1b  = (const float*)d_in[o + 10];
        bp[b].fc2w  = (const float*)d_in[o + 11];
        bp[b].fc2b  = (const float*)d_in[o + 12];
    }

    // workspace carve (256B aligned)
    size_t off = 0;
    auto carve = [&](size_t bytes) -> void* {
        void* p = (char*)d_ws + off;
        off += (bytes + 255) & ~(size_t)255;
        return p;
    };
    __bf16 *wqkv[2], *wproj[2], *wfc1[2], *wfc2[2];
    for (int b = 0; b < 2; b++) {
        wqkv[b]  = (__bf16*)carve((size_t)768 * 256 * 2);
        wproj[b] = (__bf16*)carve((size_t)256 * 256 * 2);
        wfc1[b]  = (__bf16*)carve((size_t)1024 * 256 * 2);
        wfc2[b]  = (__bf16*)carve((size_t)256 * 1024 * 2);
    }
    __bf16* ybuf   = (__bf16*)carve((size_t)TPC * 256 * 2);
    __bf16* qkvbuf = (__bf16*)carve((size_t)TPC * 768 * 2);
    __bf16* hbuf   = (__bf16*)carve((size_t)TPC * 1024 * 2);
    float*  x1buf  = (float*)carve((size_t)TPC * 256 * 4);
    float*  xb1    = (float*)carve((size_t)TOK * 256 * 4);

    // weight conversion (transpose to (N,K) bf16)
    for (int b = 0; b < 2; b++) {
        wconv_kernel<<<(256 * 768 + 255) / 256, 256, 0, stream>>>(bp[b].qkvw, wqkv[b], 256, 768);
        wconv_kernel<<<(256 * 256 + 255) / 256, 256, 0, stream>>>(bp[b].projw, wproj[b], 256, 256);
        wconv_kernel<<<(256 * 1024 + 255) / 256, 256, 0, stream>>>(bp[b].fc1w, wfc1[b], 256, 1024);
        wconv_kernel<<<(1024 * 256 + 255) / 256, 256, 0, stream>>>(bp[b].fc2w, wfc2[b], 1024, 256);
    }

    for (int ch = 0; ch < NCH; ch++) {
        const float* xin  = x + (size_t)ch * TPC * C_;
        float* xb1c  = xb1 + (size_t)ch * TPC * C_;
        float* xoutc = (float*)d_out + (size_t)ch * TPC * C_;
        for (int b = 0; b < 2; b++) {
            const float* bin  = (b == 0) ? xin : (const float*)xb1c;
            float*       bout = (b == 0) ? xb1c : xoutc;
            int shift = (b == 0) ? 0 : 3;

            // LN1 -> bf16 tokens
            ln_kernel<<<TPC / 8, 256, 0, stream>>>(bin, bp[b].ln1g, bp[b].ln1b, ybuf, TPC);
            // QKV GEMM (bias -> bf16)
            gemm_kernel<0><<<dim3(768 / 64, TPC / 128), 256, 0, stream>>>(
                ybuf, wqkv[b], bp[b].qkvb, nullptr, qkvbuf, TPC, 768, 256);
            // windowed attention -> reuse ybuf as attention output
            attn_kernel<<<dim3(IMGC * 16, HEADS_), 128, 0, stream>>>(
                qkvbuf, bp[b].btab, ybuf, shift);
            // proj GEMM + residual -> f32 x1
            gemm_kernel<1><<<dim3(256 / 64, TPC / 128), 256, 0, stream>>>(
                ybuf, wproj[b], bp[b].projb, bin, x1buf, TPC, 256, 256);
            // LN2 -> bf16
            ln_kernel<<<TPC / 8, 256, 0, stream>>>(x1buf, bp[b].ln2g, bp[b].ln2b, ybuf, TPC);
            // fc1 GEMM + GELU -> bf16
            gemm_kernel<2><<<dim3(1024 / 64, TPC / 128), 256, 0, stream>>>(
                ybuf, wfc1[b], bp[b].fc1b, nullptr, hbuf, TPC, 1024, 256);
            // fc2 GEMM + residual -> f32 block output
            gemm_kernel<1><<<dim3(256 / 64, TPC / 128), 256, 0, stream>>>(
                hbuf, wfc2[b], bp[b].fc2b, x1buf, bout, TPC, 256, 1024);
        }
    }
}
